// CachedDecoderAttention_14577119002800
// MI455X (gfx1250) — matrix operand order repaired and checked by reading, not compiled, not run on hardware
//
#include <hip/hip_runtime.h>
#include <hip/hip_bf16.h>

// ---------------------------------------------------------------------------
// CDNA5 / gfx1250 fused decoder attention (bf16 WMMA, fp32 accumulate)
//   N_HEADS=16, D_MODEL=1024, D_HEAD=64, BATCH=2, SEQ=2048  (BP = 4096 rows)
// ---------------------------------------------------------------------------

typedef __attribute__((ext_vector_type(16))) __bf16         v16bf;
typedef __attribute__((ext_vector_type(8)))  float          v8f;
typedef __attribute__((ext_vector_type(8)))  unsigned short v8u16;

union Frag {
    v16bf bf;
    v8u16 u[2];
};

__device__ __forceinline__ unsigned short f2bf(float f) {
    union { float f; unsigned int u; } cv; cv.f = f;
    unsigned int r = (cv.u + 0x7FFFu + ((cv.u >> 16) & 1u)) >> 16;   // RNE
    return (unsigned short)r;
}

// A-fragment (16x32 bf16): lane holds row m=lane&15; per ISA layout lanes 0-15
// hold K 0-7 / 16-23, lanes 16-31 hold K 8-15 / 24-31 (packed pairs).
__device__ __forceinline__ v16bf load_fragA(const unsigned short* p, int ld, int lane) {
    const int m = lane & 15, half = lane >> 4;
    Frag f;
    f.u[0] = *(const v8u16*)(p + (size_t)m * ld + 8 * half);
    f.u[1] = *(const v8u16*)(p + (size_t)m * ld + 16 + 8 * half);
    return f.bf;
}

// B-fragment (32x16 bf16) loaded from *transposed* storage Bt[n][K] (row = n):
// lanes 0-15 hold K 0-7 / 16-23, lanes 16-31 hold K 8-15 / 24-31 (matches A).
__device__ __forceinline__ v16bf load_fragB(const unsigned short* p, int ld, int lane) {
    const int n = lane & 15, half = lane >> 4;
    Frag f;
    f.u[0] = *(const v8u16*)(p + (size_t)n * ld + 8 * half);
    f.u[1] = *(const v8u16*)(p + (size_t)n * ld + 16 + 8 * half);
    return f.bf;
}

__device__ __forceinline__ v8f wmma_bf16(v16bf a, v16bf b, v8f c) {
    return __builtin_amdgcn_wmma_f32_16x16x32_bf16(false, a, false, b, (short)0, c, false, false);
}

// ---------------------------------------------------------------------------
// Packing kernels (fp32 -> bf16, with weight transposes for contiguous frags)
// ---------------------------------------------------------------------------
__global__ void cvt_f32_bf16_kernel(const float* __restrict__ src,
                                    unsigned short* __restrict__ dst, int n) {
    int i = blockIdx.x * blockDim.x + threadIdx.x;
    if (i < n) dst[i] = f2bf(src[i]);
}

// W [16][1024][64] (h,d,k) -> Wt [16][64][1024] (h,k,d)
__global__ void pack_wT_kernel(const float* __restrict__ W,
                               unsigned short* __restrict__ Wt) {
    int i = blockIdx.x * blockDim.x + threadIdx.x;   // 16*64*1024
    int d = i & 1023;
    int n = (i >> 10) & 63;
    int h = i >> 16;
    Wt[i] = f2bf(W[((size_t)h * 1024 + d) * 64 + n]);
}

// O [1024_hk][1024_d] -> Ot [1024_d][1024_hk]
__global__ void pack_oT_kernel(const float* __restrict__ O,
                               unsigned short* __restrict__ Ot) {
    int i = blockIdx.x * blockDim.x + threadIdx.x;   // 1024*1024
    int hk = i & 1023;
    int d  = i >> 10;
    Ot[i] = f2bf(O[(size_t)hk * 1024 + d]);
}

// ---------------------------------------------------------------------------
// QKV projection: X[4096x1024] x Wt_h[64x1024]^T -> 16x64 tile per wave.
// q,k stored [B][H][S][64]; v stored transposed [B][H][64][S].
// ---------------------------------------------------------------------------
__global__ void __launch_bounds__(128) qkv_gemm_kernel(
    const unsigned short* __restrict__ xb,
    const unsigned short* __restrict__ wqT,
    const unsigned short* __restrict__ wkT,
    const unsigned short* __restrict__ wvT,
    const float* __restrict__ qbias,
    const float* __restrict__ kbias,
    const float* __restrict__ vbias,
    unsigned short* __restrict__ qo,
    unsigned short* __restrict__ ko,
    unsigned short* __restrict__ vTo) {
    const int wid   = (blockIdx.x * blockDim.x + threadIdx.x) >> 5;
    const int lane  = threadIdx.x & 31;
    const int mtile = wid & 255;          // 256 tiles of 16 rows over BP=4096
    const int h     = (wid >> 8) & 15;
    const int which = wid >> 12;          // 0=Q 1=K 2=V

    const unsigned short* W = (which == 0) ? wqT : ((which == 1) ? wkT : wvT);
    const float* bias       = (which == 0) ? qbias : ((which == 1) ? kbias : vbias);
    const unsigned short* A = xb + (size_t)(mtile * 16) * 1024;
    const unsigned short* B = W + (size_t)h * 64 * 1024;

    v8f acc[4] = {};
    for (int kk = 0; kk < 1024; kk += 32) {
        v16bf a = load_fragA(A + kk, 1024, lane);
#pragma unroll
        for (int n = 0; n < 4; ++n) {
            v16bf b = load_fragB(B + (size_t)(n * 16) * 1024 + kk, 1024, lane);
            acc[n] = wmma_bf16(a, b, acc[n]);
        }
    }

    const int half = lane >> 4, nlo = lane & 15;
    const int bb    = (mtile * 16) >> 11;     // batch index
    const int srow0 = (mtile * 16) & 2047;    // row within sequence

    if (which < 2) {
        unsigned short* dst =
            ((which == 0) ? qo : ko) + (((size_t)bb * 16 + h) * 2048 + srow0) * 64;
#pragma unroll
        for (int n = 0; n < 4; ++n) {
            float bv = bias[h * 64 + n * 16 + nlo];
#pragma unroll
            for (int r = 0; r < 8; ++r) {
                int M = r + 8 * half;
                dst[(size_t)M * 64 + n * 16 + nlo] = f2bf(acc[n][r] + bv);
            }
        }
    } else {
        unsigned short* dst = vTo + (((size_t)bb * 16 + h) * 64) * 2048;
#pragma unroll
        for (int n = 0; n < 4; ++n) {
            float bv = bias[h * 64 + n * 16 + nlo];
            int   c  = n * 16 + nlo;
            v8u16 pk;
#pragma unroll
            for (int r = 0; r < 8; ++r) pk[r] = f2bf(acc[n][r] + bv);
            *(v8u16*)(dst + (size_t)c * 2048 + srow0 + 8 * half) = pk;  // 16B store
        }
    }
}

// ---------------------------------------------------------------------------
// Flash attention: one wave owns 16 query rows, streams 32-key blocks.
// ---------------------------------------------------------------------------
__global__ void __launch_bounds__(128) attn_kernel(
    const unsigned short* __restrict__ qb,
    const unsigned short* __restrict__ kb,
    const unsigned short* __restrict__ vT,
    unsigned short* __restrict__ zb) {
    __shared__ __align__(16) unsigned short lds[4][16 * 32];   // per-wave P tile

    const int widb = threadIdx.x >> 5;
    const int lane = threadIdx.x & 31;
    const int wid  = blockIdx.x * 4 + widb;
    const int qt   = wid & 127;           // 128 query tiles per (b,h)
    const int h    = (wid >> 7) & 15;
    const int b    = wid >> 11;
    const int q0   = qt * 16;

    const unsigned short* qbase = qb + (((size_t)b * 16 + h) * 2048 + q0) * 64;
    const unsigned short* kbase = kb + (((size_t)b * 16 + h) * 2048) * 64;
    const unsigned short* vbase = vT + (((size_t)b * 16 + h) * 64) * 2048;

    const v16bf a_lo = load_fragA(qbase, 64, lane);        // K = 0..31
    const v16bf a_hi = load_fragA(qbase + 32, 64, lane);   // K = 32..63

    const int half = lane >> 4, nlo = lane & 15;
    float mi[8], li[8];
    v8f   zacc[4] = {};
#pragma unroll
    for (int r = 0; r < 8; ++r) { mi[r] = -1e30f; li[r] = 0.f; }

    const int nkb = ((q0 + 15) >> 5) + 1;   // causal: key blocks of 32
    for (int kbk = 0; kbk < nkb; ++kbk) {
        const int key0 = kbk * 32;
        float s[2][8];
#pragma unroll
        for (int t = 0; t < 2; ++t) {
            const unsigned short* kp = kbase + (size_t)(key0 + t * 16) * 64;
            v16bf b0 = load_fragB(kp, 64, lane);        // K rows as B (already "transposed")
            v16bf b1 = load_fragB(kp + 32, 64, lane);
            v8f c = {};
            c = wmma_bf16(a_lo, b0, c);
            c = wmma_bf16(a_hi, b1, c);
#pragma unroll
            for (int r = 0; r < 8; ++r) {
                int   qrow = q0 + r + 8 * half;
                int   kcol = key0 + t * 16 + nlo;
                float v    = c[r] * 0.125f;             // 1/sqrt(64)
                s[t][r]    = (kcol > qrow) ? -1e30f : v;
            }
        }
        // Online softmax: row r lives across 16 lanes of one half.
        float p0[8], p1[8];
#pragma unroll
        for (int r = 0; r < 8; ++r) {
            float lm = fmaxf(s[0][r], s[1][r]);
#pragma unroll
            for (int off = 1; off < 16; off <<= 1) lm = fmaxf(lm, __shfl_xor(lm, off, 32));
            float mn    = fmaxf(mi[r], lm);
            float alpha = __expf(mi[r] - mn);
            float e0    = __expf(s[0][r] - mn);
            float e1    = __expf(s[1][r] - mn);
            float rs    = e0 + e1;
#pragma unroll
            for (int off = 1; off < 16; off <<= 1) rs += __shfl_xor(rs, off, 32);
            li[r] = li[r] * alpha + rs;
            mi[r] = mn;
            p0[r] = e0; p1[r] = e1;
#pragma unroll
            for (int n = 0; n < 4; ++n) zacc[n][r] *= alpha;
        }
        // Transpose P (C-layout) -> A-layout through per-wave LDS slice.
        unsigned short* sl = lds[widb];
#pragma unroll
        for (int r = 0; r < 8; ++r) {
            int M = r + 8 * half;
            sl[M * 32 + nlo]      = f2bf(p0[r]);
            sl[M * 32 + 16 + nlo] = f2bf(p1[r]);
        }
        __asm__ volatile("s_wait_dscnt 0" ::: "memory");   // RAW on LDS within wave
        v16bf pf = load_fragA(sl, 32, lane);
#pragma unroll
        for (int n = 0; n < 4; ++n) {
            v16bf vb = load_fragB(vbase + (size_t)(n * 16) * 2048 + key0, 2048, lane);
            zacc[n] = wmma_bf16(pf, vb, zacc[n]);
        }
    }

    // Normalize and emit z as bf16 into [B*S][1024] (cols h*64..h*64+63).
    unsigned short* dst = zb + ((size_t)b * 2048 + q0) * 1024 + h * 64;
#pragma unroll
    for (int n = 0; n < 4; ++n) {
#pragma unroll
        for (int r = 0; r < 8; ++r) {
            int M = r + 8 * half;
            dst[(size_t)M * 1024 + n * 16 + nlo] = f2bf(zacc[n][r] / li[r]);
        }
    }
}

// ---------------------------------------------------------------------------
// Output projection: Z[4096x1024] x O^T[1024x1024] + Ob -> fp32 out
// ---------------------------------------------------------------------------
__global__ void __launch_bounds__(128) out_gemm_kernel(
    const unsigned short* __restrict__ zb,
    const unsigned short* __restrict__ oT,
    const float* __restrict__ obias,
    float* __restrict__ out) {
    const int wid   = (blockIdx.x * blockDim.x + threadIdx.x) >> 5;
    const int lane  = threadIdx.x & 31;
    const int mtile = wid & 255;   // 16-row tiles over 4096
    const int nt    = wid >> 8;    // 16 tiles of 64 cols over 1024

    const unsigned short* A = zb + (size_t)(mtile * 16) * 1024;
    v8f acc[4] = {};
    for (int kk = 0; kk < 1024; kk += 32) {
        v16bf a = load_fragA(A + kk, 1024, lane);
#pragma unroll
        for (int n = 0; n < 4; ++n) {
            v16bf b = load_fragB(oT + (size_t)(nt * 64 + n * 16) * 1024 + kk, 1024, lane);
            acc[n] = wmma_bf16(a, b, acc[n]);
        }
    }
    const int half = lane >> 4, nlo = lane & 15;
    float* dst = out + (size_t)(mtile * 16) * 1024 + nt * 64;
#pragma unroll
    for (int n = 0; n < 4; ++n) {
        float bv = obias[nt * 64 + n * 16 + nlo];
#pragma unroll
        for (int r = 0; r < 8; ++r)
            dst[(size_t)(r + 8 * half) * 1024 + n * 16 + nlo] = acc[n][r] + bv;
    }
}

// ---------------------------------------------------------------------------
// Host launcher
// ---------------------------------------------------------------------------
extern "C" void kernel_launch(void* const* d_in, const int* in_sizes, int n_in,
                              void* d_out, int out_size, void* d_ws, size_t ws_size,
                              hipStream_t stream) {
    const float* x   = (const float*)d_in[0];
    const float* Qs  = (const float*)d_in[1];
    const float* Qbs = (const float*)d_in[2];
    const float* Ks  = (const float*)d_in[3];
    const float* Kbs = (const float*)d_in[4];
    const float* Vs  = (const float*)d_in[5];
    const float* Vbs = (const float*)d_in[6];
    const float* O   = (const float*)d_in[7];
    const float* Ob  = (const float*)d_in[8];
    float* out = (float*)d_out;

    char* ws = (char*)d_ws;
    const size_t MB = 1u << 20;
    unsigned short* xb  = (unsigned short*)(ws + 0 * MB);   // 8 MiB  [4096][1024] bf16
    unsigned short* wqT = (unsigned short*)(ws + 8 * MB);   // 2 MiB  [16][64][1024]
    unsigned short* wkT = (unsigned short*)(ws + 10 * MB);  // 2 MiB
    unsigned short* wvT = (unsigned short*)(ws + 12 * MB);  // 2 MiB
    unsigned short* oT  = (unsigned short*)(ws + 14 * MB);  // 2 MiB  [1024][1024]
    unsigned short* qb  = (unsigned short*)(ws + 16 * MB);  // 8 MiB  [2][16][2048][64]
    unsigned short* kbf = (unsigned short*)(ws + 24 * MB);  // 8 MiB
    unsigned short* vT  = (unsigned short*)(ws + 32 * MB);  // 8 MiB  [2][16][64][2048]
    unsigned short* zb  = (unsigned short*)(ws + 40 * MB);  // 8 MiB  [4096][1024]

    // Pack / convert
    cvt_f32_bf16_kernel<<<(4096 * 1024) / 256, 256, 0, stream>>>(x, xb, 4096 * 1024);
    pack_wT_kernel<<<(16 * 64 * 1024) / 256, 256, 0, stream>>>(Qs, wqT);
    pack_wT_kernel<<<(16 * 64 * 1024) / 256, 256, 0, stream>>>(Ks, wkT);
    pack_wT_kernel<<<(16 * 64 * 1024) / 256, 256, 0, stream>>>(Vs, wvT);
    pack_oT_kernel<<<(1024 * 1024) / 256, 256, 0, stream>>>(O, oT);

    // QKV projection: 3 * 16 heads * 256 row-tiles = 12288 waves, 4 waves/block
    qkv_gemm_kernel<<<3072, 128, 0, stream>>>(xb, wqT, wkT, wvT, Qbs, Kbs, Vbs,
                                              qb, kbf, vT);

    // Attention: 2 * 16 * 128 = 4096 waves
    attn_kernel<<<1024, 128, 0, stream>>>(qb, kbf, vT, zb);

    // Output projection: 256 * 16 = 4096 waves
    out_gemm_kernel<<<1024, 128, 0, stream>>>(zb, oT, Ob, out);
}
